// Center_31568009625975
// MI455X (gfx1250) — matrix operand order repaired and checked by reading, not compiled, not run on hardware
//
#include <hip/hip_runtime.h>
#include <hip/hip_bf16.h>

typedef __attribute__((ext_vector_type(16))) _Float16 v16h;
typedef __attribute__((ext_vector_type(8)))  _Float16 v8h;
typedef __attribute__((ext_vector_type(8)))  float    v8f;

#define NB 2
#define CH 64
#define NY 80
#define NX 80
#define NC 80
#define NPIX (NY*NX)          // 6400
#define KPB  (NC*NPIX)        // 512000 rows per batch
#define NO   (5+NC)           // 85
#define BLK256 256
#define BLOCKS_PER_BATCH (KPB/BLK256)   // 2000

// ---------------- Stage 1: three 3x3 conv heads as implicit GEMM on WMMA ----
// Tiles: head(3) x batch(2) x ntile(4, 16 out-ch each) x mtile(400, 16 px each)
__global__ __launch_bounds__(256)
void conv3x3_wmma(const float* __restrict__ x,
                  const float* __restrict__ w_hm, const float* __restrict__ b_hm,
                  const float* __restrict__ w_wh, const float* __restrict__ b_wh,
                  const float* __restrict__ w_rg, const float* __restrict__ b_rg,
                  _Float16* __restrict__ y1h)
{
    const int wave = blockIdx.x * (blockDim.x >> 5) + (threadIdx.x >> 5);
    if (wave >= 3 * NB * 4 * 400) return;           // wave-uniform guard
    int mtile = wave % 400;
    int t     = wave / 400;
    int ntile = t % 4;  t /= 4;
    int b     = t % NB;
    int h     = t / NB;

    const float* w1 = (h == 0) ? w_hm : (h == 1) ? w_wh : w_rg;
    const float* b1 = (h == 0) ? b_hm : (h == 1) ? b_wh : b_rg;

    const int lane = threadIdx.x & 31;
    const int ml   = lane & 15;          // M for A-frag, N for B/C-frag
    const int g    = lane >> 4;          // lane group

    const int p  = mtile * 16 + ml;      // pixel for A rows (same row y for all 16)
    const int y  = p / NX;
    const int xc = p % NX;
    const float* xb = x + (size_t)b * CH * NPIX;
    const int nout = ntile * 16 + ml;

    v8f acc = {};
    #pragma unroll
    for (int ky = 0; ky < 3; ++ky) {
        const int yy  = y + ky - 1;
        const int yyc = yy < 0 ? 0 : (yy > NY - 1 ? NY - 1 : yy);
        const bool yok = (unsigned)yy < (unsigned)NY;
        #pragma unroll
        for (int kx = 0; kx < 3; ++kx) {
            const int xx  = xc + kx - 1;
            const int xxc = xx < 0 ? 0 : (xx > NX - 1 ? NX - 1 : xx);
            const bool ok = yok && ((unsigned)xx < (unsigned)NX);
            // clamped, always-in-bounds base; pad handled by select after load
            const float* ap = xb + yyc * NX + xxc;
            const float* wp = w1 + (size_t)nout * CH * 9 + ky * 3 + kx;
            #pragma unroll
            for (int s = 0; s < 2; ++s) {
                const int cbase = 32 * s;
                v16h a, bb;
                // A fragment: A[m][c] = x[b][c][yy][xx(m)], zero-padded via select
                #pragma unroll
                for (int i = 0; i < 16; ++i) {
                    const int c = cbase + ((i < 8) ? (g * 8 + i) : (8 + g * 8 + i));
                    const float v = ap[c * NPIX];          // unconditional load
                    a[i] = (_Float16)(ok ? v : 0.0f);      // branch-free pad
                }
                // B fragment: B[c][n] = w1[n][c][ky][kx]
                #pragma unroll
                for (int i = 0; i < 16; ++i) {
                    const int c = cbase + ((i < 8) ? (g * 8 + i) : (8 + g * 8 + i));
                    bb[i] = (_Float16)wp[c * 9];
                }
                acc = __builtin_amdgcn_wmma_f32_16x16x32_f16(
                        false, a, false, bb, (short)0, acc, false, false);
            }
        }
    }
    // bias + ReLU + store pixel-major f16: y1h[(h*NB+b)*NPIX + p][c]
    const float bias = b1[nout];
    #pragma unroll
    for (int r = 0; r < 8; ++r) {
        const int m  = r + 8 * g;
        float v = acc[r] + bias;
        v = v > 0.0f ? v : 0.0f;
        const int pp = mtile * 16 + m;
        y1h[((size_t)(h * NB + b) * NPIX + pp) * CH + nout] = (_Float16)v;
    }
}

// ---------------- Stage 2a: hm 1x1 conv (64->80) + sigmoid on WMMA ----------
// Tiles: batch(2) x ntile(5) x mtile(400) = 4000 waves
__global__ __launch_bounds__(256)
void conv1x1_hm_wmma(const _Float16* __restrict__ y1h,
                     const float* __restrict__ w2, const float* __restrict__ b2,
                     float* __restrict__ hm_ws)
{
    const int wave = blockIdx.x * (blockDim.x >> 5) + (threadIdx.x >> 5);
    if (wave >= NB * 5 * 400) return;               // wave-uniform guard
    int mtile = wave % 400;
    int t     = wave / 400;
    int ntile = t % 5;
    int b     = t / 5;

    const int lane = threadIdx.x & 31;
    const int ml   = lane & 15;
    const int g    = lane >> 4;
    const int p    = mtile * 16 + ml;
    const int nout = ntile * 16 + ml;

    // A rows are channel-contiguous: load as 16B v8h chunks
    const v8h*    av = (const v8h*)(y1h + ((size_t)(0 * NB + b) * NPIX + p) * CH);
    const float4* wv = (const float4*)(w2 + (size_t)nout * CH);

    v8f acc = {};
    #pragma unroll
    for (int s = 0; s < 2; ++s) {
        const int cbase = 32 * s;
        v16h a, bb;
        // halves 0..7  -> c = cbase + 16g .. +7     => chunk (cbase/8 + 2g)
        // halves 8..15 -> c = cbase + 16g + 8 .. => chunk (cbase/8 + 2g + 1)
        const v8h c0 = av[cbase / 8 + 2 * g];
        const v8h c1 = av[cbase / 8 + 2 * g + 1];
        #pragma unroll
        for (int i = 0; i < 8; ++i) { a[i] = c0[i]; a[i + 8] = c1[i]; }
        // B: 16 consecutive floats starting at c = cbase + 16g
        const int wb = (cbase + g * 16) / 4;
        const float4 w0 = wv[wb + 0], w1v = wv[wb + 1],
                     w2v = wv[wb + 2], w3v = wv[wb + 3];
        bb[0]  = (_Float16)w0.x;  bb[1]  = (_Float16)w0.y;
        bb[2]  = (_Float16)w0.z;  bb[3]  = (_Float16)w0.w;
        bb[4]  = (_Float16)w1v.x; bb[5]  = (_Float16)w1v.y;
        bb[6]  = (_Float16)w1v.z; bb[7]  = (_Float16)w1v.w;
        bb[8]  = (_Float16)w2v.x; bb[9]  = (_Float16)w2v.y;
        bb[10] = (_Float16)w2v.z; bb[11] = (_Float16)w2v.w;
        bb[12] = (_Float16)w3v.x; bb[13] = (_Float16)w3v.y;
        bb[14] = (_Float16)w3v.z; bb[15] = (_Float16)w3v.w;
        acc = __builtin_amdgcn_wmma_f32_16x16x32_f16(
                false, a, false, bb, (short)0, acc, false, false);
    }
    const float bias = b2[nout];
    #pragma unroll
    for (int r = 0; r < 8; ++r) {
        const int m = r + 8 * g;
        float v = acc[r] + bias;
        v = 1.0f / (1.0f + __expf(-v));             // sigmoid
        const int pp = mtile * 16 + m;
        hm_ws[((size_t)b * NC + nout) * NPIX + pp] = v;
    }
}

// ---------------- Stage 2b: wh/reg 1x1 convs (64->2 each) + xywh ------------
__global__ __launch_bounds__(256)
void whreg_xywh(const _Float16* __restrict__ y1h,
                const float* __restrict__ wh_w2, const float* __restrict__ wh_b2,
                const float* __restrict__ rg_w2, const float* __restrict__ rg_b2,
                const float* __restrict__ offsets,
                float* __restrict__ xywh)
{
    const int tid = blockIdx.x * blockDim.x + threadIdx.x;
    if (tid >= NB * NPIX) return;
    const int b = tid / NPIX, p = tid % NPIX;
    const _Float16* Aw = y1h + ((size_t)(1 * NB + b) * NPIX + p) * CH;
    const _Float16* Ar = y1h + ((size_t)(2 * NB + b) * NPIX + p) * CH;
    float wh0 = wh_b2[0], wh1 = wh_b2[1], rg0 = rg_b2[0], rg1 = rg_b2[1];
    #pragma unroll 8
    for (int c = 0; c < CH; ++c) {
        const float aw = (float)Aw[c], ar = (float)Ar[c];
        wh0 += aw * wh_w2[0 * CH + c];
        wh1 += aw * wh_w2[1 * CH + c];
        rg0 += ar * rg_w2[0 * CH + c];
        rg1 += ar * rg_w2[1 * CH + c];
    }
    wh0 = fmaxf(wh0, 0.0f); wh1 = fmaxf(wh1, 0.0f);
    rg0 = fmaxf(rg0, 0.0f); rg1 = fmaxf(rg1, 0.0f);
    const float offx = offsets[b * 3 + 1] * 2.0f;
    const float offy = offsets[b * 3 + 2] * 2.0f;
    const int yv = p / NX, xv = p % NX;
    // xyxy->xywh collapses: cx = ctr*STRIDE, w = wh*STRIDE
    float4 o;
    o.x = ((float)xv + offx + rg0) * 4.0f;
    o.y = ((float)yv + offy + rg1) * 4.0f;
    o.z = wh0 * 4.0f;
    o.w = wh1 * 4.0f;
    *(float4*)(xywh + (size_t)(b * NPIX + p) * 4) = o;
}

// ---------------- Stage 3a: 3x3 peak mask + per-block counts ----------------
// -inf-padded 3x3 window max == replicate-clamped window max (clamped
// neighbors land on pixels already inside the valid window): unconditional loads.
__global__ __launch_bounds__(256)
void maxima_count(const float* __restrict__ hm_ws,
                  unsigned char* __restrict__ flags,
                  int* __restrict__ blockSums)
{
    const int tid = blockIdx.x * BLK256 + threadIdx.x;   // 0 .. NB*KPB-1
    const int b = tid / KPB, k = tid % KPB;
    const int c = k / NPIX, r = k % NPIX;
    const int y = r / NX, x = r % NX;
    const float* plane = hm_ws + ((size_t)b * NC + c) * NPIX;
    const int ym = y > 0 ? y - 1 : 0,  yp = y < NY - 1 ? y + 1 : NY - 1;
    const int xm = x > 0 ? x - 1 : 0,  xp = x < NX - 1 ? x + 1 : NX - 1;
    const float v = plane[y * NX + x];
    float mx = v;
    mx = fmaxf(mx, plane[ym * NX + xm]);
    mx = fmaxf(mx, plane[ym * NX + x ]);
    mx = fmaxf(mx, plane[ym * NX + xp]);
    mx = fmaxf(mx, plane[y  * NX + xm]);
    mx = fmaxf(mx, plane[y  * NX + xp]);
    mx = fmaxf(mx, plane[yp * NX + xm]);
    mx = fmaxf(mx, plane[yp * NX + x ]);
    mx = fmaxf(mx, plane[yp * NX + xp]);
    const int f = (mx == v) ? 1 : 0;
    flags[tid] = (unsigned char)f;

    __shared__ int sh[BLK256];
    sh[threadIdx.x] = f; __syncthreads();
    #pragma unroll
    for (int off = BLK256 / 2; off > 0; off >>= 1) {
        if (threadIdx.x < off) sh[threadIdx.x] += sh[threadIdx.x + off];
        __syncthreads();
    }
    if (threadIdx.x == 0) blockSums[blockIdx.x] = sh[0];
}

// ---------------- Stage 3b: serial exclusive scan + per-batch totals --------
__global__ void scan_blocks(const int* __restrict__ blockSums,
                            int* __restrict__ blockOffs,
                            int* __restrict__ counts)
{
    if (threadIdx.x != 0) return;
    const int b = blockIdx.x;                 // one block per batch
    int acc = 0;
    for (int i = 0; i < BLOCKS_PER_BATCH; ++i) {
        blockOffs[b * BLOCKS_PER_BATCH + i] = acc;
        acc += blockSums[b * BLOCKS_PER_BATCH + i];
    }
    counts[b] = acc;                          // total maxima in batch b
}

// ---------------- Stage 3c: compact maxima indices (stable order) -----------
__global__ __launch_bounds__(256)
void compact_idx(const unsigned char* __restrict__ flags,
                 const int* __restrict__ blockOffs,
                 int* __restrict__ idx)
{
    const int tid = blockIdx.x * BLK256 + threadIdx.x;
    const int b = tid / KPB, k = tid % KPB;
    const int f = flags[tid];

    __shared__ int sh[BLK256];
    sh[threadIdx.x] = f; __syncthreads();
    for (int off = 1; off < BLK256; off <<= 1) {
        const int t = (threadIdx.x >= off) ? sh[threadIdx.x - off] : 0;
        __syncthreads();
        sh[threadIdx.x] += t;
        __syncthreads();
    }
    const int excl = sh[threadIdx.x] - f;
    if (f) {
        const int rank = blockOffs[blockIdx.x] + excl;
        idx[b * KPB + rank] = k;              // source row for dst rank
    }
}

// ---------------- Stage 3d: single coalesced pass over the 348MB output -----
// One wave per output row (85 floats). Lanes cover cols {lane, lane+32, lane+64};
// rows >= count[b] are all-zero. Non-temporal stores: 348MB > L2, pure stream.
__global__ __launch_bounds__(256)
void write_out(const int* __restrict__ idx,
               const int* __restrict__ counts,
               const float* __restrict__ xywh,
               const float* __restrict__ hm_ws,
               float* __restrict__ out)
{
    const int wave = blockIdx.x * (blockDim.x >> 5) + (threadIdx.x >> 5);
    const int row  = wave;                    // 0 .. NB*KPB-1
    const int b    = row / KPB;
    const int rank = row % KPB;
    const int lane = threadIdx.x & 31;

    float v0 = 0.0f, v1 = 0.0f, v2 = 0.0f;
    if (rank < counts[b]) {                   // wave-uniform branch
        const int k = idx[b * KPB + rank];
        const int c = k / NPIX, r = k % NPIX;
        const float4 xw = *(const float4*)(xywh + (size_t)(b * NPIX + r) * 4);
        const float score = hm_ws[((size_t)b * NC + c) * NPIX + r];
        // col = lane
        v0 = (lane == 0) ? xw.x : (lane == 1) ? xw.y :
             (lane == 2) ? xw.z : (lane == 3) ? xw.w :
             (lane == 4) ? score : (lane == 5 + c ? 1.0f : 0.0f);
        // col = lane + 32
        v1 = (lane + 32 == 5 + c) ? 1.0f : 0.0f;
        // col = lane + 64 (valid for lane < 21)
        v2 = (lane + 64 == 5 + c) ? 1.0f : 0.0f;
    }
    float* rp = out + (size_t)row * NO;
    __builtin_nontemporal_store(v0, rp + lane);
    __builtin_nontemporal_store(v1, rp + lane + 32);
    if (lane < NO - 64)                       // cols 64..84
        __builtin_nontemporal_store(v2, rp + lane + 64);
}

// ---------------- Host launcher ---------------------------------------------
extern "C" void kernel_launch(void* const* d_in, const int* in_sizes, int n_in,
                              void* d_out, int out_size, void* d_ws, size_t ws_size,
                              hipStream_t stream)
{
    (void)in_sizes; (void)n_in; (void)ws_size; (void)out_size;
    const float* x       = (const float*)d_in[0];
    const float* offsets = (const float*)d_in[1];
    const float* hm_w1   = (const float*)d_in[2];
    const float* hm_b1   = (const float*)d_in[3];
    const float* hm_w2   = (const float*)d_in[4];
    const float* hm_b2   = (const float*)d_in[5];
    const float* wh_w1   = (const float*)d_in[6];
    const float* wh_b1   = (const float*)d_in[7];
    const float* wh_w2   = (const float*)d_in[8];
    const float* wh_b2   = (const float*)d_in[9];
    const float* rg_w1   = (const float*)d_in[10];
    const float* rg_b1   = (const float*)d_in[11];
    const float* rg_w2   = (const float*)d_in[12];
    const float* rg_b2   = (const float*)d_in[13];

    // Workspace layout (bytes)
    char* ws = (char*)d_ws;
    _Float16*      y1h   = (_Float16*)     (ws + 0);          // 4,915,200 B
    float*         hm_ws = (float*)        (ws + 4915200);    // 4,096,000 B
    float*         xywh  = (float*)        (ws + 9011200);    //   204,800 B
    unsigned char* flags = (unsigned char*)(ws + 9216000);    // 1,024,000 B
    int*           bsums = (int*)          (ws + 10240000);   //    16,000 B
    int*           boffs = (int*)          (ws + 10256000);   //    16,000 B
    int*           cnts  = (int*)          (ws + 10272000);   //       256 B (padded)
    int*           idx   = (int*)          (ws + 10272256);   // 4,096,000 B
    float*         out   = (float*)d_out;

    conv3x3_wmma   <<<1200, 256, 0, stream>>>(x, hm_w1, hm_b1, wh_w1, wh_b1,
                                              rg_w1, rg_b1, y1h);
    conv1x1_hm_wmma<<<500,  256, 0, stream>>>(y1h, hm_w2, hm_b2, hm_ws);
    whreg_xywh     <<<50,   256, 0, stream>>>(y1h, wh_w2, wh_b2, rg_w2, rg_b2,
                                              offsets, xywh);
    maxima_count   <<<NB * BLOCKS_PER_BATCH, 256, 0, stream>>>(hm_ws, flags, bsums);
    scan_blocks    <<<NB, 32, 0, stream>>>(bsums, boffs, cnts);
    compact_idx    <<<NB * BLOCKS_PER_BATCH, 256, 0, stream>>>(flags, boffs, idx);
    // one wave per output row: NB*KPB rows, 8 waves per 256-thread block
    write_out      <<<NB * KPB / 8, 256, 0, stream>>>(idx, cnts, xywh, hm_ws, out);
}
